// ADJSCC4_more_21285857919596
// MI455X (gfx1250) — matrix operand order repaired and checked by reading, not compiled, not run on hardware
//
#include <hip/hip_runtime.h>
#include <math.h>

typedef __attribute__((ext_vector_type(16))) _Float16 v16h;
typedef __attribute__((ext_vector_type(8)))  _Float16 v8h;
typedef __attribute__((ext_vector_type(8)))  float    v8f;

// ---------------------------------------------------------------- utilities

__global__ void k_zero(float* __restrict__ p, int n) {
  int i = blockIdx.x * blockDim.x + threadIdx.x;
  if (i < n) p[i] = 0.f;
}

// elementwise f32 -> f16 (input (B,HW,C) is already NHWC)
__global__ void k_cvt_f16(const float* __restrict__ src, _Float16* __restrict__ dst, int n) {
  int i = blockIdx.x * blockDim.x + threadIdx.x;
  if (i < n) dst[i] = (_Float16)src[i];
}

// pack weights to f16 [kh][kw][co][ci]  (per-lane-contiguous B operand)
// convt=0: src OIHW  ((co*Cin+ci)*K+kh)*K+kw
// convt=1: src [Cin,Cout,k,k]  ((ci*Cout+co)*K+kh)*K+kw
__global__ void k_pack_w(const float* __restrict__ src, _Float16* __restrict__ dst,
                         int Cin, int Cout, int K, int convt) {
  int i = blockIdx.x * blockDim.x + threadIdx.x;
  int n = Cin * Cout * K * K;
  if (i >= n) return;
  int ci  = i % Cin;
  int t   = i / Cin;
  int co  = t % Cout;
  int tap = t / Cout;
  int kh = tap / K, kw = tap - kh * K;
  float v = convt ? src[(((size_t)ci * Cout + co) * K + kh) * K + kw]
                  : src[(((size_t)co * Cin + ci) * K + kh) * K + kw];
  dst[i] = (_Float16)v;
}

// ------------------------------------------------- WMMA forward convolution
// NHWC f16 activations, packed f16 weights. One wave: 16 pixels x 32 couts.
__global__ void k_conv_wmma(const _Float16* __restrict__ in, const _Float16* __restrict__ wpk,
                            const float* __restrict__ bias, const _Float16* __restrict__ zpad,
                            float* __restrict__ out,
                            int B, int Cin, int Cout, int Hin, int Win,
                            int Hout, int Wout, int K, int S) {
  const int lane = threadIdx.x & 31;
  const int wid  = (blockIdx.x * blockDim.x + threadIdx.x) >> 5;
  const int npix = Hout * Wout;
  const int ptiles = (npix + 15) >> 4;
  const int ntiles = Cout >> 5;
  const int total  = B * ptiles * ntiles;
  if (wid >= total) return;                 // wave-uniform: EXEC stays all-ones

  const int b  = wid / (ptiles * ntiles);
  const int r0 = wid - b * (ptiles * ntiles);
  const int pt = r0 / ntiles;
  const int nt = r0 - pt * ntiles;

  const int mrow = lane & 15;               // A-row / B-column / D-column
  const int hi   = lane >> 4;
  int op = pt * 16 + mrow;
  const bool mv = (op < npix);
  if (!mv) op = 0;
  const int oh = op / Wout;
  const int ow = op - oh * Wout;
  const int co0 = nt * 32 + mrow;

  const _Float16* inb = in + (size_t)b * Hin * Win * Cin;

  v8f acc0 = {0.f, 0.f, 0.f, 0.f, 0.f, 0.f, 0.f, 0.f};
  v8f acc1 = {0.f, 0.f, 0.f, 0.f, 0.f, 0.f, 0.f, 0.f};

  for (int kh = 0; kh < K; ++kh) {
    const int ih = oh * S + kh;
    for (int kw = 0; kw < K; ++kw) {
      const int iw = ow * S + kw;
      const _Float16* srcp = mv ? inb + ((size_t)ih * Win + iw) * Cin : zpad;
      const _Float16* wtap = wpk + ((size_t)(kh * K + kw) * Cout + co0) * Cin;
      for (int kb = 0; kb < Cin; kb += 32) {
        // A: lanes0-15 K in kb+0..15 (hi=0), kb+16..31 for hi=1 -> 2x b128
        const v8h a0 = *(const v8h*)(srcp + kb + hi * 16);
        const v8h a1 = *(const v8h*)(srcp + kb + hi * 16 + 8);
        const v16h av = __builtin_shufflevector(a0, a1, 0, 1, 2, 3, 4, 5, 6, 7,
                                                8, 9, 10, 11, 12, 13, 14, 15);
        // B: lane's 16 K values contiguous in packed layout
        const v16h bv0 = *(const v16h*)(wtap + kb + hi * 16);
        const v16h bv1 = *(const v16h*)(wtap + (size_t)16 * Cin + kb + hi * 16);
        acc0 = __builtin_amdgcn_wmma_f32_16x16x32_f16(false, av, false, bv0,
                                                      (short)0, acc0, false, false);
        acc1 = __builtin_amdgcn_wmma_f32_16x16x32_f16(false, av, false, bv1,
                                                      (short)0, acc1, false, false);
      }
    }
  }
  const float bb0 = bias[co0];
  const float bb1 = bias[co0 + 16];
#pragma unroll
  for (int r = 0; r < 8; ++r) {
    const int opix = pt * 16 + r + hi * 8;  // C/D: M = r + 8*hi, N = lane&15
    if (opix < npix) {
      out[((size_t)b * Cout + co0) * npix + opix]      = acc0[r] + bb0;
      out[((size_t)b * Cout + co0 + 16) * npix + opix] = acc1[r] + bb1;
    }
  }
}

// --------------------------------------------- WMMA transposed convolution
// out[oh,ow] += in[ih,iw]*w[ci,co,kh,kw] where oh = ih*S - P + kh
__global__ void k_convt_wmma(const _Float16* __restrict__ in, const _Float16* __restrict__ wpk,
                             const float* __restrict__ bias, const _Float16* __restrict__ zpad,
                             float* __restrict__ out,
                             int B, int Cin, int Cout, int Hin, int Win,
                             int Hout, int Wout, int K, int S, int P) {
  const int lane = threadIdx.x & 31;
  const int wid  = (blockIdx.x * blockDim.x + threadIdx.x) >> 5;
  const int npix = Hout * Wout;
  const int ptiles = (npix + 15) >> 4;
  const int ntiles = Cout >> 5;
  const int total  = B * ptiles * ntiles;
  if (wid >= total) return;

  const int b  = wid / (ptiles * ntiles);
  const int r0 = wid - b * (ptiles * ntiles);
  const int pt = r0 / ntiles;
  const int nt = r0 - pt * ntiles;

  const int mrow = lane & 15;
  const int hi   = lane >> 4;
  int op = pt * 16 + mrow;
  const bool mv = (op < npix);
  if (!mv) op = 0;
  const int oh = op / Wout;
  const int ow = op - oh * Wout;
  const int co0 = nt * 32 + mrow;

  const _Float16* inb = in + (size_t)b * Hin * Win * Cin;

  v8f acc0 = {0.f, 0.f, 0.f, 0.f, 0.f, 0.f, 0.f, 0.f};
  v8f acc1 = {0.f, 0.f, 0.f, 0.f, 0.f, 0.f, 0.f, 0.f};

  for (int kh = 0; kh < K; ++kh) {
    const int nh = oh + P - kh;
    int ih = nh / S;
    const bool vh = mv && (nh >= 0) && (ih * S == nh) && (ih < Hin);
    if (!vh) ih = 0;
    for (int kw = 0; kw < K; ++kw) {
      const int nw = ow + P - kw;
      int iw = nw / S;
      const bool vw = vh && (nw >= 0) && (iw * S == nw) && (iw < Win);
      if (!vw) iw = 0;
      const _Float16* srcp = vw ? inb + ((size_t)ih * Win + iw) * Cin : zpad;
      const _Float16* wtap = wpk + ((size_t)(kh * K + kw) * Cout + co0) * Cin;
      for (int kb = 0; kb < Cin; kb += 32) {
        const v8h a0 = *(const v8h*)(srcp + kb + hi * 16);
        const v8h a1 = *(const v8h*)(srcp + kb + hi * 16 + 8);
        const v16h av = __builtin_shufflevector(a0, a1, 0, 1, 2, 3, 4, 5, 6, 7,
                                                8, 9, 10, 11, 12, 13, 14, 15);
        const v16h bv0 = *(const v16h*)(wtap + kb + hi * 16);
        const v16h bv1 = *(const v16h*)(wtap + (size_t)16 * Cin + kb + hi * 16);
        acc0 = __builtin_amdgcn_wmma_f32_16x16x32_f16(false, av, false, bv0,
                                                      (short)0, acc0, false, false);
        acc1 = __builtin_amdgcn_wmma_f32_16x16x32_f16(false, av, false, bv1,
                                                      (short)0, acc1, false, false);
      }
    }
  }
  const float bb0 = bias[co0];
  const float bb1 = bias[co0 + 16];
#pragma unroll
  for (int r = 0; r < 8; ++r) {
    const int opix = pt * 16 + r + hi * 8;
    if (opix < npix) {
      out[((size_t)b * Cout + co0) * npix + opix]      = acc0[r] + bb0;
      out[((size_t)b * Cout + co0 + 16) * npix + opix] = acc1[r] + bb1;
    }
  }
}

// ---------------------------------------------------------- GDN + PReLU (LDS-tiled)
#define GDN_G 64
__global__ void k_gdn_prelu(const float* __restrict__ x, const float* __restrict__ gamma,
                            const float* __restrict__ beta, const float* __restrict__ alpha,
                            float* __restrict__ out, int C, int npix, int ngroups,
                            int inverse) {
  __shared__ float tile[128 * GDN_G];       // up to C=128 -> 32 KB
  const int b  = blockIdx.x / ngroups;
  const int g0 = (blockIdx.x % ngroups) * GDN_G;
  const int gs = min(GDN_G, npix - g0);
  const float* xb = x + ((size_t)b * C) * npix + g0;
  for (int idx = threadIdx.x; idx < C * GDN_G; idx += blockDim.x) {
    const int c = idx / GDN_G, p = idx - c * GDN_G;
    tile[idx] = (p < gs) ? xb[(size_t)c * npix + p] : 0.f;
  }
  __syncthreads();
  const float a = alpha[0];
  for (int idx = threadIdx.x; idx < C * GDN_G; idx += blockDim.x) {
    const int k = idx / GDN_G, p = idx - k * GDN_G;
    if (p >= gs) continue;
    const float* gk = gamma + (size_t)k * C;
    float z = beta[k];
    for (int c = 0; c < C; ++c) {
      float v = tile[c * GDN_G + p];
      z += gk[c] * v * v;
    }
    float s  = sqrtf(z);
    float xv = tile[k * GDN_G + p];
    float y  = inverse ? xv * s : xv / s;
    out[((size_t)b * C + k) * npix + g0 + p] = (y >= 0.f) ? y : a * y;
  }
}

// ------------------------------------------------- per-sample power norm
__global__ void k_sample_norm(const float* __restrict__ x, float* __restrict__ scale, int per) {
  __shared__ float sm[256];
  int b = blockIdx.x;
  const float* xb = x + (size_t)b * per;
  float s = 0.f;
  for (int i = threadIdx.x; i < per; i += 256) { float v = xb[i]; s += v * v; }
  sm[threadIdx.x] = s;
  __syncthreads();
  for (int o = 128; o > 0; o >>= 1) {
    if (threadIdx.x < o) sm[threadIdx.x] += sm[threadIdx.x + o];
    __syncthreads();
  }
  if (threadIdx.x == 0) scale[b] = 32.f / sqrtf(sm[0]);
}

// --------------------------------------------------- quantize (+outputs)
__global__ void k_quant(const float* __restrict__ x, const float* __restrict__ scale,
                        const float* __restrict__ centers, float* __restrict__ xq,
                        float* __restrict__ sent, float* __restrict__ bins,
                        float* __restrict__ partial, int B, int C, int npix) {
  __shared__ float sb[8];
  __shared__ float sm[256];
  int i = blockIdx.x * blockDim.x + threadIdx.x;
  int n = B * C * npix;
  if (threadIdx.x < 8) sb[threadIdx.x] = 0.f;
  __syncthreads();
  float h2 = 0.f;
  if (i < n) {
    int pix = i % npix;
    int t   = i / npix;
    int c   = t % C;
    int b   = t / C;
    float xv = x[i] * scale[b];
    int best = 0; float bd = 3.4e38f;
    for (int j = 0; j < 8; ++j) {
      float d = xv - centers[j];
      d *= d;
      if (d < bd) { bd = d; best = j; }
    }
    float hv = centers[best];             // forward of STE == hard value
    xq[i] = hv;
    sent[((size_t)b * npix + pix) * C + c] = hv;
    atomicAdd(&sb[best], 1.f);
    h2 = hv * hv;
  }
  sm[threadIdx.x] = h2;
  __syncthreads();
  for (int o = 128; o > 0; o >>= 1) {
    if (threadIdx.x < o) sm[threadIdx.x] += sm[threadIdx.x + o];
    __syncthreads();
  }
  if (threadIdx.x == 0) partial[blockIdx.x] = sm[0];
  if (threadIdx.x < 8) atomicAdd(&bins[threadIdx.x], sb[threadIdx.x]);
}

__global__ void k_reduce_sum(const float* __restrict__ partial, int n, float* __restrict__ dst) {
  __shared__ float sm[256];
  float s = 0.f;
  for (int i = threadIdx.x; i < n; i += 256) s += partial[i];
  sm[threadIdx.x] = s;
  __syncthreads();
  for (int o = 128; o > 0; o >>= 1) {
    if (threadIdx.x < o) sm[threadIdx.x] += sm[threadIdx.x + o];
    __syncthreads();
  }
  if (threadIdx.x == 0) dst[0] = sm[0];
}

__global__ void k_count_final(const float* __restrict__ bins, float* __restrict__ dst, float invN) {
  int j = threadIdx.x;
  if (j < 8) dst[j] = bins[j] * invN;
}

// ------------------- AWGN channel; also writes NHWC f16 decoder input
__global__ void k_noise(const float* __restrict__ xq, const float* __restrict__ noise,
                        const float* __restrict__ sumsq, const float* __restrict__ snr,
                        float* __restrict__ cwn, _Float16* __restrict__ xh,
                        int C, int npix, int n, float invN) {
  int i = blockIdx.x * blockDim.x + threadIdx.x;
  if (i >= n) return;
  float snr_lin = exp10f(snr[0] * 0.1f);
  float stdv    = sqrtf(sumsq[0] * invN / snr_lin);
  float v = xq[i] + stdv * noise[i];
  cwn[i] = v;
  int pix = i % npix;
  int t   = i / npix;
  int c   = t % C;
  int b   = t / C;
  xh[((size_t)b * npix + pix) * C + c] = (_Float16)v;
}

// ------------------------------------------------------------ AF module
__global__ void k_chan_mean(const float* __restrict__ x, float* __restrict__ t, int npix) {
  __shared__ float sm[256];
  int bc = blockIdx.x;
  const float* xb = x + (size_t)bc * npix;
  float s = 0.f;
  for (int i = threadIdx.x; i < npix; i += 256) s += xb[i];
  sm[threadIdx.x] = s;
  __syncthreads();
  for (int o = 128; o > 0; o >>= 1) {
    if (threadIdx.x < o) sm[threadIdx.x] += sm[threadIdx.x + o];
    __syncthreads();
  }
  if (threadIdx.x == 0) t[bc] = sm[0] / (float)npix;
}

__global__ void k_af_fc1(const float* __restrict__ t, const float* __restrict__ snr,
                         const float* __restrict__ w1, const float* __restrict__ b1,
                         float* __restrict__ h, int B, int C) {
  int i = blockIdx.x * blockDim.x + threadIdx.x;
  if (i >= B * C) return;
  int j = i % C;
  int b = i / C;
  const float* wr = w1 + (size_t)j * (C + 1);
  const float* tb = t + (size_t)b * C;
  float s = b1[j];
  for (int c = 0; c < C; ++c) s += wr[c] * tb[c];
  s += wr[C] * snr[0];
  h[i] = (s > 0.f) ? s : 0.f;
}

__global__ void k_af_fc2(const float* __restrict__ h, const float* __restrict__ w2,
                         const float* __restrict__ b2, float* __restrict__ g, int B, int C) {
  int i = blockIdx.x * blockDim.x + threadIdx.x;
  if (i >= B * C) return;
  int j = i % C;
  int b = i / C;
  const float* wr = w2 + (size_t)j * C;
  const float* hb = h + (size_t)b * C;
  float s = b2[j];
  for (int c = 0; c < C; ++c) s += wr[c] * hb[c];
  g[i] = 1.f / (1.f + expf(-s));
}

// scale by channel gate, emitting the next conv's NHWC f16 input directly
__global__ void k_af_scale_h(const float* __restrict__ x, const float* __restrict__ g,
                             _Float16* __restrict__ xh, int C, int npix, int n) {
  int i = blockIdx.x * blockDim.x + threadIdx.x;   // i over (b, pix, c)
  if (i >= n) return;
  int c   = i % C;
  int t   = i / C;
  int pix = t % npix;
  int b   = t / npix;
  float v = x[((size_t)b * C + c) * npix + pix] * g[b * C + c];
  xh[i] = (_Float16)v;
}

// --------------------- final: inverse GDN + sigmoid + (B,HW,C) store (LDS-tiled)
__global__ void k_final(const float* __restrict__ x, const float* __restrict__ gamma,
                        const float* __restrict__ beta, float* __restrict__ out,
                        int C, int npix, int ngroups) {
  __shared__ float tile[64 * GDN_G];        // C=64 -> 16 KB
  const int b  = blockIdx.x / ngroups;
  const int g0 = (blockIdx.x % ngroups) * GDN_G;
  const int gs = min(GDN_G, npix - g0);
  const float* xb = x + ((size_t)b * C) * npix + g0;
  for (int idx = threadIdx.x; idx < C * GDN_G; idx += blockDim.x) {
    const int c = idx / GDN_G, p = idx - c * GDN_G;
    tile[idx] = (p < gs) ? xb[(size_t)c * npix + p] : 0.f;
  }
  __syncthreads();
  for (int idx = threadIdx.x; idx < C * GDN_G; idx += blockDim.x) {
    const int k = idx / GDN_G, p = idx - k * GDN_G;
    if (p >= gs) continue;
    const float* gk = gamma + (size_t)k * C;
    float z = beta[k];
    for (int c = 0; c < C; ++c) {
      float v = tile[c * GDN_G + p];
      z += gk[c] * v * v;
    }
    float val = tile[k * GDN_G + p] * sqrtf(z);
    out[((size_t)b * npix + g0 + p) * C + k] = 1.f / (1.f + expf(-val));
  }
}

// ------------------------------------------------------------ host driver

static inline void launch_convt(const _Float16* in, const _Float16* wpk, const float* b,
                                const _Float16* zpad, float* out,
                                int B, int Cin, int Cout, int Hin, int Win,
                                int Hout, int Wout, int K, int S, int P, hipStream_t s) {
  int npix  = Hout * Wout;
  int waves = B * ((npix + 15) / 16) * (Cout / 32);
  int th    = waves * 32;
  k_convt_wmma<<<(th + 255) / 256, 256, 0, s>>>(in, wpk, b, zpad, out, B, Cin, Cout,
                                                Hin, Win, Hout, Wout, K, S, P);
}

static inline void launch_gdn(const float* x, const float* g, const float* be,
                              const float* a, float* out, int B, int C, int npix,
                              int inverse, hipStream_t s) {
  int ngroups = (npix + GDN_G - 1) / GDN_G;
  k_gdn_prelu<<<B * ngroups, 256, 0, s>>>(x, g, be, a, out, C, npix, ngroups, inverse);
}

extern "C" void kernel_launch(void* const* d_in, const int* in_sizes, int n_in,
                              void* d_out, int out_size, void* d_ws, size_t ws_size,
                              hipStream_t stream) {
  (void)in_sizes; (void)n_in; (void)out_size; (void)ws_size;

  const float* inp   = (const float*)d_in[0];
  const float* SNR   = (const float*)d_in[1];
  const float* cent  = (const float*)d_in[2];
  const float* noise = (const float*)d_in[3];
  const float* fl1_w = (const float*)d_in[4];
  const float* fl1_b = (const float*)d_in[5];
  const float* fl1_g = (const float*)d_in[6];
  const float* fl1_be= (const float*)d_in[7];
  const float* fl1_a = (const float*)d_in[8];
  const float* tc4_w = (const float*)d_in[45];
  const float* tc4_b = (const float*)d_in[46];
  const float* g4_g  = (const float*)d_in[47];
  const float* g4_be = (const float*)d_in[48];

  float* out = (float*)d_out;
  float* ws  = (float*)d_ws;

  const int B = 32;
  const size_t OFF_SENT = 0;
  const size_t OFF_CWN  = 746496;
  const size_t OFF_RECV = 1492992;
  const size_t OFF_COUNT= 25380864;
  const size_t OFF_XQ   = 25380872;

  // workspace layout (float offsets)
  const size_t STATS_F = 32768;
  const size_t SLAB_F  = 37850112;           // max activation: 32*96*111*111
  const size_t C5_F    = 23887872;           // 32*64*108*108
  float* stats   = ws;
  float* tbuf    = stats + 64;               // 4096
  float* hbuf    = stats + 64 + 4096;        // 4096
  float* gbuf    = stats + 64 + 8192;        // 4096
  float* partial = stats + 16384;            // 2916
  _Float16* zpad = (_Float16*)(stats + 24576); // 256 zero halves (stays zero)
  float* slabA   = ws + STATS_F;             // f32 activations
  float* slabB   = slabA + SLAB_F;           // f32 conv outputs
  float* slabC   = slabB + SLAB_F;           // tc4 f32 output
  _Float16* halfA = (_Float16*)(slabC + C5_F);          // f16 NHWC conv inputs
  _Float16* wpk   = halfA + SLAB_F;                     // packed f16 weights
  const size_t WH_FL1 = 0,      WH_D1 = 32768,  WH_D2 = 51200,
               WH_D3  = 106496, WH_D4 = 217088, WH_TC4 = 327680;

  // 0) zero stats region (also establishes zpad zeros)
  k_zero<<<(int)((STATS_F + 255) / 256), 256, 0, stream>>>(stats, (int)STATS_F);

  // 1) pack all weights to f16 [tap][co][ci]
  { int n = 64 * 32 * 16;
    k_pack_w<<<(n + 255) / 256, 256, 0, stream>>>(fl1_w, wpk + WH_FL1, 64, 32, 4, 0); }
  { int n = 32 * 64 * 9;
    k_pack_w<<<(n + 255) / 256, 256, 0, stream>>>((const float*)d_in[9],  wpk + WH_D1, 32, 64, 3, 1); }
  { int n = 64 * 96 * 9;
    k_pack_w<<<(n + 255) / 256, 256, 0, stream>>>((const float*)d_in[18], wpk + WH_D2, 64, 96, 3, 1); }
  { int n = 96 * 128 * 9;
    k_pack_w<<<(n + 255) / 256, 256, 0, stream>>>((const float*)d_in[27], wpk + WH_D3, 96, 128, 3, 1); }
  { int n = 128 * 96 * 9;
    k_pack_w<<<(n + 255) / 256, 256, 0, stream>>>((const float*)d_in[36], wpk + WH_D4, 128, 96, 3, 1); }
  { int n = 96 * 64 * 4;
    k_pack_w<<<(n + 255) / 256, 256, 0, stream>>>(tc4_w, wpk + WH_TC4, 96, 64, 2, 1); }

  // 2) input (B,HW,C) is already NHWC: just convert to f16
  { int n = B * 3136 * 64;
    k_cvt_f16<<<(n + 255) / 256, 256, 0, stream>>>(inp, halfA, n); }

  // 3) encoder conv 64->32, k4 s2 : halfA -> slabB
  { int waves = B * ((729 + 15) / 16) * (32 / 32);
    int th = waves * 32;
    k_conv_wmma<<<(th + 255) / 256, 256, 0, stream>>>(halfA, wpk + WH_FL1, fl1_b, zpad,
                                                      slabB, B, 64, 32, 56, 56, 27, 27, 4, 2); }

  // 4) GDN (forward) + PReLU : slabB -> slabA
  launch_gdn(slabB, fl1_g, fl1_be, fl1_a, slabA, B, 32, 729, 0, stream);

  // 5) per-sample power normalization scale -> stats[0..31]
  k_sample_norm<<<B, 256, 0, stream>>>(slabA, stats, 32 * 729);

  // 6) quantize -> x_q slot + codeSent slot; bins + partial sums
  k_quant<<<746496 / 256, 256, 0, stream>>>(slabA, stats, cent, out + OFF_XQ,
                                            out + OFF_SENT, stats + 40, partial,
                                            B, 32, 729);
  k_reduce_sum<<<1, 256, 0, stream>>>(partial, 746496 / 256, stats + 32);
  k_count_final<<<1, 8, 0, stream>>>(stats + 40, out + OFF_COUNT, 1.f / 746496.f);

  // 7) AWGN channel -> codeWithNoise slot + NHWC f16 decoder input
  k_noise<<<(746496 + 255) / 256, 256, 0, stream>>>(out + OFF_XQ, noise, stats + 32,
                                                    SNR, out + OFF_CWN, halfA,
                                                    32, 729, 746496, 1.f / 746496.f);

  // 8) decoder stages: convT -> iGDN+PReLU -> AF (emits next f16 NHWC input)
  struct StageDef { int wi; size_t wh; int Cin, Cout, Hin, Win, Hout, Wout, K, S, P; };
  const StageDef st[4] = {
    { 9, WH_D1,  32,  64, 27, 27,  27,  27, 3, 1, 1},
    {18, WH_D2,  64,  96, 27, 27,  29,  29, 3, 1, 0},
    {27, WH_D3,  96, 128, 29, 29,  57,  57, 3, 2, 1},
    {36, WH_D4, 128,  96, 57, 57, 111, 111, 3, 2, 2},
  };
  for (int si = 0; si < 4; ++si) {
    const StageDef& S_ = st[si];
    const float* gg = (const float*)d_in[S_.wi + 2];
    const float* be = (const float*)d_in[S_.wi + 3];
    const float* aa = (const float*)d_in[S_.wi + 4];
    const float* w1 = (const float*)d_in[S_.wi + 5];
    const float* b1 = (const float*)d_in[S_.wi + 6];
    const float* w2 = (const float*)d_in[S_.wi + 7];
    const float* b2 = (const float*)d_in[S_.wi + 8];
    const float* bb = (const float*)d_in[S_.wi + 1];
    int npix = S_.Hout * S_.Wout;
    launch_convt(halfA, wpk + S_.wh, bb, zpad, slabB, B, S_.Cin, S_.Cout,
                 S_.Hin, S_.Win, S_.Hout, S_.Wout, S_.K, S_.S, S_.P, stream);
    launch_gdn(slabB, gg, be, aa, slabA, B, S_.Cout, npix, 1, stream);
    k_chan_mean<<<B * S_.Cout, 256, 0, stream>>>(slabA, tbuf, npix);
    int bc = B * S_.Cout;
    k_af_fc1<<<(bc + 255) / 256, 256, 0, stream>>>(tbuf, SNR, w1, b1, hbuf, B, S_.Cout);
    k_af_fc2<<<(bc + 255) / 256, 256, 0, stream>>>(hbuf, w2, b2, gbuf, B, S_.Cout);
    int n = bc * npix;
    k_af_scale_h<<<(n + 255) / 256, 256, 0, stream>>>(slabA, gbuf, halfA, S_.Cout, npix, n);
  }

  // 9) tc4 convT 96->64 k2 s1 p2 : halfA -> slabC
  launch_convt(halfA, wpk + WH_TC4, tc4_b, zpad, slabC, B, 96, 64,
               111, 111, 108, 108, 2, 1, 2, stream);

  // 10) inverse GDN + sigmoid + transpose -> codeReceived slot
  { int ngroups = (11664 + GDN_G - 1) / GDN_G;
    k_final<<<B * ngroups, 256, 0, stream>>>(slabC, g4_g, g4_be,
                                             out + OFF_RECV, 64, 11664, ngroups); }
}
